// HybridAttentionDecoder_15925738734030
// MI455X (gfx1250) — matrix operand order repaired and checked by reading, not compiled, not run on hardware
//
#include <hip/hip_runtime.h>
#include <hip/hip_bf16.h>

// ---------------------------------------------------------------------------
// Types / helpers for CDNA5 WMMA bf16 (wave32)
// ---------------------------------------------------------------------------
typedef __attribute__((ext_vector_type(16))) __bf16 v16bf;
typedef __attribute__((ext_vector_type(8)))  float  v8f;
typedef unsigned short u16b;

#define WMMA_BF16(a, b, c) \
  __builtin_amdgcn_wmma_f32_16x16x32_bf16(false, (a), false, (b), (short)0, (c), false, false)

// Native CDNA5 V_TANH_F32 if the toolchain exposes it; fall back to libm.
#if defined(__has_builtin)
#if __has_builtin(__builtin_amdgcn_tanhf)
#define FAST_TANH(x) __builtin_amdgcn_tanhf(x)
#elif __has_builtin(__builtin_amdgcn_tanh_f32)
#define FAST_TANH(x) __builtin_amdgcn_tanh_f32(x)
#else
#define FAST_TANH(x) tanhf(x)
#endif
#else
#define FAST_TANH(x) tanhf(x)
#endif

__device__ __forceinline__ float bf2f(u16b x) {
  return __uint_as_float(((unsigned)x) << 16);
}
__device__ __forceinline__ u16b f2bf(float f) {
  unsigned u = __float_as_uint(f);
  u += 0x7FFFu + ((u >> 16) & 1u);   // round-to-nearest-even
  return (u16b)(u >> 16);
}
__device__ __forceinline__ unsigned pack2bf(float lo, float hi) {
  return (unsigned)f2bf(lo) | ((unsigned)f2bf(hi) << 16);
}

// A-operand: 16x32 bf16 tile from a row-major matrix (row stride ld elements).
// ISA 7.12.2 layout: lane half h, m=lane&15 -> row=row0+m,
//   element e<8 : K = 8h + e           (contiguous)
//   element e>=8: K = 16 + 8h + (e-8)  (contiguous)
__device__ __forceinline__ v16bf load_a(const u16b* base, size_t ld, int row0, int k0, int lane) {
  const int h = lane >> 4, m = lane & 15;
  const __bf16* p0 = (const __bf16*)(base + (size_t)(row0 + m) * ld + (size_t)(k0 + 8 * h));
  const __bf16* p1 = p0 + 16;
  v16bf a;
#pragma unroll
  for (int e = 0; e < 8; ++e) a[e] = p0[e];
#pragma unroll
  for (int e = 0; e < 8; ++e) a[8 + e] = p1[e];
  return a;
}

// B-operand: 32x16 bf16 tile representing W^T, loaded from row-major W [N,K].
// lane half h, m -> column n = n0+m, element e<8 -> K = k0 + 8h + e, e>=8 -> K = k0 + 16 + 8h + (e-8).
__device__ __forceinline__ v16bf load_b(const u16b* W, size_t ld, int n0, int k0, int lane) {
  const int h = lane >> 4, m = lane & 15;
  const __bf16* p = (const __bf16*)(W + (size_t)(n0 + m) * ld + (size_t)(k0 + 8 * h));
  v16bf b;
#pragma unroll
  for (int e = 0; e < 8; ++e) b[e] = p[e];
#pragma unroll
  for (int e = 0; e < 8; ++e) b[8 + e] = p[16 + e];
  return b;
}

// Problem constants
#define S 512
#define VOCAB 1000
#define Bsz 64
#define TDEC 128
#define TFEAT 1024
#define KW 31
#define PADW 15
#define WPLD 1056  // 15 + 1024 + 17 padded alignment row

// ---------------------------------------------------------------------------
// Setup kernels
// ---------------------------------------------------------------------------
// n must be a multiple of 4 (all call sites are)
__global__ void k_f2bf4(const float* __restrict__ src, unsigned* __restrict__ dst, int n4) {
  int i = blockIdx.x * blockDim.x + threadIdx.x;
  if (i >= n4) return;
  float4 f = ((const float4*)src)[i];
  dst[2 * i]     = pack2bf(f.x, f.y);
  dst[2 * i + 1] = pack2bf(f.z, f.w);
}

__global__ void k_zero(unsigned* __restrict__ dst, int ndw) {
  int i = blockIdx.x * blockDim.x + threadIdx.x;
  if (i < ndw) dst[i] = 0u;
}

// conv_w [512,1,31] -> bf16 [512,32] (K padded with zero)
__global__ void k_cvt_conv(const float* __restrict__ cw, u16b* __restrict__ dst) {
  int i = blockIdx.x * blockDim.x + threadIdx.x;
  if (i >= S * 32) return;
  int s = i >> 5, k = i & 31;
  dst[i] = (k < KW) ? f2bf(cw[s * KW + k]) : (u16b)0;
}

// Wo [1000,1024] -> bf16 [1008,1024] zero-padded rows
__global__ void k_cvt_wo(const float* __restrict__ Wo, u16b* __restrict__ dst) {
  int i = blockIdx.x * blockDim.x + threadIdx.x;
  if (i >= 1008 * 1024) return;
  int r = i >> 10, c = i & 1023;
  dst[i] = (r < VOCAB) ? f2bf(Wo[r * 1024 + c]) : (u16b)0;
}

// emb gather: dst[b,t,s] = bf16(emb[input[b,t], s])
__global__ void k_emb(const int* __restrict__ input, const float* __restrict__ emb,
                      u16b* __restrict__ dst) {
  int i = blockIdx.x * blockDim.x + threadIdx.x;
  if (i >= Bsz * TDEC * S) return;
  int s = i & (S - 1);
  int bt = i >> 9;
  int tok = input[bt];
  dst[i] = f2bf(emb[(size_t)tok * S + s]);
}

// ---------------------------------------------------------------------------
// keys = features @ Wk^T + bk   (M=65536, N=512, K=512) -> bf16
// grid (4096, 32), block 32 (one wave per 16x16 tile)
// ---------------------------------------------------------------------------
__global__ void k_keys(const u16b* __restrict__ featbf, const u16b* __restrict__ Wkbf,
                       const float* __restrict__ bk, u16b* __restrict__ keysbf) {
  const int lane = threadIdx.x;
  const int row0 = blockIdx.x * 16;
  const int n0 = blockIdx.y * 16;
  v8f c = {0.f, 0.f, 0.f, 0.f, 0.f, 0.f, 0.f, 0.f};
  for (int k0 = 0; k0 < S; k0 += 32) {
    v16bf a = load_a(featbf, S, row0, k0, lane);
    v16bf b = load_b(Wkbf, S, n0, k0, lane);
    c = WMMA_BF16(a, b, c);
  }
  const int h = lane >> 4, m = lane & 15;
  const float bias = bk[n0 + m];
#pragma unroll
  for (int r = 0; r < 8; ++r) {
    int row = row0 + r + 8 * h;
    keysbf[(size_t)row * S + n0 + m] = f2bf(c[r] + bias);
  }
}

// ---------------------------------------------------------------------------
// Fused GRU step: gi = [emb_t|ctx] @ W_ih^T, gh = h @ W_hh^T, gate math,
// h_new written f32 + bf16.   grid (4, 32), block 32.
// ---------------------------------------------------------------------------
__global__ void k_gru(const u16b* __restrict__ embbf, int t, const u16b* __restrict__ ctxbf,
                      const u16b* __restrict__ hbf_in, const float* __restrict__ hF_in,
                      const u16b* __restrict__ Wihbf, const u16b* __restrict__ Whhbf,
                      const float* __restrict__ b_ih, const float* __restrict__ b_hh,
                      float* __restrict__ hF_out, u16b* __restrict__ hbf_out) {
  const int lane = threadIdx.x;
  const int row0 = blockIdx.x * 16;  // batch rows
  const int n0 = blockIdx.y * 16;    // gate column tile within [0,512)
  v8f acc[6];
#pragma unroll
  for (int g = 0; g < 6; ++g) acc[g] = (v8f){0.f, 0.f, 0.f, 0.f, 0.f, 0.f, 0.f, 0.f};

  // input part: x = [emb_t (512) | ctx (512)], K = 1024
  const u16b* embt = embbf + (size_t)t * S;  // row stride TDEC*S between batches
  for (int k0 = 0; k0 < 2 * S; k0 += 32) {
    v16bf a = (k0 < S) ? load_a(embt, (size_t)TDEC * S, row0, k0, lane)
                       : load_a(ctxbf, S, row0, k0 - S, lane);
#pragma unroll
    for (int g = 0; g < 3; ++g) {
      v16bf b = load_b(Wihbf, 2 * S, g * S + n0, k0, lane);
      acc[g] = WMMA_BF16(a, b, acc[g]);
    }
  }
  // hidden part: K = 512
  for (int k0 = 0; k0 < S; k0 += 32) {
    v16bf a = load_a(hbf_in, S, row0, k0, lane);
#pragma unroll
    for (int g = 0; g < 3; ++g) {
      v16bf b = load_b(Whhbf, S, g * S + n0, k0, lane);
      acc[3 + g] = WMMA_BF16(a, b, acc[3 + g]);
    }
  }

  const int h = lane >> 4, m = lane & 15;
  const int col = n0 + m;
  const float bir = b_ih[col], biz = b_ih[S + col], bin = b_ih[2 * S + col];
  const float bhr = b_hh[col], bhz = b_hh[S + col], bhn = b_hh[2 * S + col];
#pragma unroll
  for (int r = 0; r < 8; ++r) {
    int row = row0 + r + 8 * h;
    float ir = acc[0][r] + bir, iz = acc[1][r] + biz, in_ = acc[2][r] + bin;
    float hr = acc[3][r] + bhr, hz = acc[4][r] + bhz, hn = acc[5][r] + bhn;
    float rr = 1.f / (1.f + __expf(-(ir + hr)));
    float zz = 1.f / (1.f + __expf(-(iz + hz)));
    float nn = FAST_TANH(in_ + rr * hn);
    float hprev = hF_in[(size_t)row * S + col];
    float hnew = (1.f - zz) * nn + zz * hprev;
    hF_out[(size_t)row * S + col] = hnew;
    hbf_out[(size_t)row * S + col] = f2bf(hnew);
  }
}

// ---------------------------------------------------------------------------
// q = h_new @ Wq^T + bq   (M=64, N=512, K=512) -> f32.  grid (4,32), block 32.
// ---------------------------------------------------------------------------
__global__ void k_q(const u16b* __restrict__ hbf, const u16b* __restrict__ Wqbf,
                    const float* __restrict__ bq, float* __restrict__ qF) {
  const int lane = threadIdx.x;
  const int row0 = blockIdx.x * 16;
  const int n0 = blockIdx.y * 16;
  v8f c = {0.f, 0.f, 0.f, 0.f, 0.f, 0.f, 0.f, 0.f};
  for (int k0 = 0; k0 < S; k0 += 32) {
    v16bf a = load_a(hbf, S, row0, k0, lane);
    v16bf b = load_b(Wqbf, S, n0, k0, lane);
    c = WMMA_BF16(a, b, c);
  }
  const int h = lane >> 4, m = lane & 15;
  const float bias = bq[n0 + m];
#pragma unroll
  for (int r = 0; r < 8; ++r)
    qF[(size_t)(row0 + r + 8 * h) * S + n0 + m] = c[r] + bias;
}

// ---------------------------------------------------------------------------
// Attention scores. One wave per (b, tf-tile of 16).
// Location conv expressed as a single K=32 WMMA per s-tile:
//   A[tf][k] = w_prev[b, tf0+tf+k-15] (from zero-padded buffer, row stride 1)
//   B[k][s]  = conv_w[s][k]           (bf16 [512,32] row-major)
// score[tf] = sum_s tanh(q[b,s] + conv_b[s] + keys[b,tf,s] + loc[tf,s]) * v[s]
// ---------------------------------------------------------------------------
__global__ void k_scores(const u16b* __restrict__ wppad, const u16b* __restrict__ convwbf,
                         const float* __restrict__ conv_b, const float* __restrict__ qF,
                         const u16b* __restrict__ keysbf, const float* __restrict__ vvec,
                         float* __restrict__ scores) {
  const int lane = threadIdx.x;
  const int b = blockIdx.x;
  const int tf0 = blockIdx.y * 16;
  const int h = lane >> 4, m = lane & 15;

  // prefetch this wave's 16 key rows (16 x 1KB block, rows 1KB apart)
  {
    const u16b* kbase = keysbf + ((size_t)b * TFEAT + tf0 + m) * S + h * 256;
    __builtin_prefetch(kbase, 0, 3);
  }

  // im2col A tile (overlapping rows => row stride 1); same for all s-tiles
  const u16b* wp = wppad + (size_t)b * WPLD + tf0;
  v16bf a = load_a(wp, 1, 0, 0, lane);

  float part[8] = {0.f, 0.f, 0.f, 0.f, 0.f, 0.f, 0.f, 0.f};
  for (int s0 = 0; s0 < S; s0 += 16) {
    v16bf bmat = load_b(convwbf, 32, s0, 0, lane);
    v8f c = {0.f, 0.f, 0.f, 0.f, 0.f, 0.f, 0.f, 0.f};
    c = WMMA_BF16(a, bmat, c);

    const int s = s0 + m;
    const float qs = qF[(size_t)b * S + s] + conv_b[s];
    const float vs = vvec[s];
    const u16b* kp = keysbf + ((size_t)b * TFEAT + tf0) * S + s;
#pragma unroll
    for (int r = 0; r < 8; ++r) {
      int trow = r + 8 * h;
      float e = FAST_TANH(qs + bf2f(kp[(size_t)trow * S]) + c[r]);
      part[r] += e * vs;
    }
  }
  // reduce over the 16 lanes of each half (masks 1..8 stay within the half)
#pragma unroll
  for (int r = 0; r < 8; ++r) {
    float p = part[r];
    for (int mask = 1; mask <= 8; mask <<= 1) p += __shfl_xor(p, mask, 32);
    if (m == 0) scores[(size_t)b * TFEAT + tf0 + r + 8 * h] = p;
  }
}

// ---------------------------------------------------------------------------
// softmax per batch row; writes f32 into d_out's alignment region, plus
// bf16 copies (wbf for ctx, wppad interior for next step's location conv).
// ---------------------------------------------------------------------------
__global__ void k_softmax(const float* __restrict__ scores, float* __restrict__ w_out,
                          u16b* __restrict__ wbf, u16b* __restrict__ wppad, int t) {
  const int b = blockIdx.x;
  const int tid = threadIdx.x;  // 256
  __shared__ float red[256];
  const float* sc = scores + (size_t)b * TFEAT;

  float mx = -1e30f;
  for (int i = tid; i < TFEAT; i += 256) mx = fmaxf(mx, sc[i]);
  red[tid] = mx;
  __syncthreads();
  for (int off = 128; off; off >>= 1) {
    if (tid < off) red[tid] = fmaxf(red[tid], red[tid + off]);
    __syncthreads();
  }
  mx = red[0];
  __syncthreads();

  float sum = 0.f;
  for (int i = tid; i < TFEAT; i += 256) sum += __expf(sc[i] - mx);
  red[tid] = sum;
  __syncthreads();
  for (int off = 128; off; off >>= 1) {
    if (tid < off) red[tid] += red[tid + off];
    __syncthreads();
  }
  const float inv = 1.f / red[0];

  float* wo = w_out + ((size_t)b * TDEC + t) * TFEAT;
  for (int i = tid; i < TFEAT; i += 256) {
    float w = __expf(sc[i] - mx) * inv;
    wo[i] = w;
    u16b wb = f2bf(w);
    wbf[(size_t)b * TFEAT + i] = wb;
    wppad[(size_t)b * WPLD + PADW + i] = wb;
  }
}

// ---------------------------------------------------------------------------
// ctx[b,d] = sum_t w[b,t] * features[b,t,d]  (bf16 streamed, 2 channels/thread)
// grid (64,2), block 128: thread covers d2 = {2d, 2d+1}.
// ---------------------------------------------------------------------------
__global__ void k_ctx(const u16b* __restrict__ wbf, const u16b* __restrict__ featbf,
                      u16b* __restrict__ ctxbf) {
  const int b = blockIdx.x;
  const int dpair = blockIdx.y * 128 + threadIdx.x;  // 0..255
  const u16b* w = wbf + (size_t)b * TFEAT;
  const unsigned* f = (const unsigned*)(featbf + (size_t)b * TFEAT * S) + dpair;
  float acc0 = 0.f, acc1 = 0.f;
  for (int tt = 0; tt < TFEAT; ++tt) {
    unsigned pk = f[(size_t)tt * (S / 2)];
    float wv = bf2f(w[tt]);
    acc0 += wv * __uint_as_float(pk << 16);
    acc1 += wv * __uint_as_float(pk & 0xFFFF0000u);
  }
  ((unsigned*)ctxbf)[(size_t)b * (S / 2) + dpair] = pack2bf(acc0, acc1);
}

// ---------------------------------------------------------------------------
// logits = [h|ctx] @ Wo^T + bo -> d_out outs[b,t,:].  grid (4,63), block 32.
// ---------------------------------------------------------------------------
__global__ void k_logits(const u16b* __restrict__ hbf, const u16b* __restrict__ ctxbf,
                         const u16b* __restrict__ Wobf, const float* __restrict__ bo,
                         float* __restrict__ outs, int t) {
  const int lane = threadIdx.x;
  const int row0 = blockIdx.x * 16;
  const int n0 = blockIdx.y * 16;
  v8f c = {0.f, 0.f, 0.f, 0.f, 0.f, 0.f, 0.f, 0.f};
  for (int k0 = 0; k0 < 2 * S; k0 += 32) {
    v16bf a = (k0 < S) ? load_a(hbf, S, row0, k0, lane)
                       : load_a(ctxbf, S, row0, k0 - S, lane);
    v16bf b = load_b(Wobf, 2 * S, n0, k0, lane);
    c = WMMA_BF16(a, b, c);
  }
  const int h = lane >> 4, m = lane & 15;
  const int col = n0 + m;
  if (col < VOCAB) {
    const float bias = bo[col];
#pragma unroll
    for (int r = 0; r < 8; ++r) {
      int row = row0 + r + 8 * h;
      outs[((size_t)row * TDEC + t) * VOCAB + col] = c[r] + bias;
    }
  }
}

// ---------------------------------------------------------------------------
// host side
// ---------------------------------------------------------------------------
extern "C" void kernel_launch(void* const* d_in, const int* in_sizes, int n_in,
                              void* d_out, int out_size, void* d_ws, size_t ws_size,
                              hipStream_t stream) {
  (void)in_sizes; (void)n_in; (void)out_size; (void)ws_size;
  const int*   input    = (const int*)d_in[0];
  const float* features = (const float*)d_in[1];
  const float* emb      = (const float*)d_in[2];
  const float* W_ih     = (const float*)d_in[3];
  const float* W_hh     = (const float*)d_in[4];
  const float* b_ih     = (const float*)d_in[5];
  const float* b_hh     = (const float*)d_in[6];
  const float* Wq       = (const float*)d_in[7];
  const float* bq       = (const float*)d_in[8];
  const float* Wk       = (const float*)d_in[9];
  const float* bk       = (const float*)d_in[10];
  const float* conv_w   = (const float*)d_in[11];
  const float* conv_b   = (const float*)d_in[12];
  const float* vvec     = (const float*)d_in[13];
  const float* Wo       = (const float*)d_in[14];
  const float* bo       = (const float*)d_in[15];

  float* outs  = (float*)d_out;                          // [64,128,1000]
  float* wsout = outs + (size_t)Bsz * TDEC * VOCAB;      // [64,128,1024]

  // workspace carve-out
  char* p = (char*)d_ws;
  auto take = [&](size_t elems, size_t esz) -> void* {
    void* r = (void*)p;
    p += ((elems * esz) + 255) & ~(size_t)255;
    return r;
  };
  const size_t N_FEAT = (size_t)Bsz * TFEAT * S;   // 33.5M
  u16b*  featbf  = (u16b*)take(N_FEAT, 2);
  u16b*  keysbf  = (u16b*)take(N_FEAT, 2);
  u16b*  embbf   = (u16b*)take((size_t)Bsz * TDEC * S, 2);
  u16b*  Wihbf   = (u16b*)take((size_t)3 * S * 2 * S, 2);
  u16b*  Whhbf   = (u16b*)take((size_t)3 * S * S, 2);
  u16b*  Wqbf    = (u16b*)take((size_t)S * S, 2);
  u16b*  Wkbf    = (u16b*)take((size_t)S * S, 2);
  u16b*  Wobf    = (u16b*)take((size_t)1008 * 1024, 2);
  u16b*  convwbf = (u16b*)take((size_t)S * 32, 2);
  float* hF0     = (float*)take((size_t)Bsz * S, 4);
  float* hF1     = (float*)take((size_t)Bsz * S, 4);
  u16b*  hbf0    = (u16b*)take((size_t)Bsz * S, 2);
  u16b*  hbf1    = (u16b*)take((size_t)Bsz * S, 2);
  u16b*  ctxbf   = (u16b*)take((size_t)Bsz * S, 2);
  float* qF      = (float*)take((size_t)Bsz * S, 4);
  float* scoresF = (float*)take((size_t)Bsz * TFEAT, 4);
  u16b*  wbf     = (u16b*)take((size_t)Bsz * TFEAT, 2);
  u16b*  wppad   = (u16b*)take((size_t)Bsz * WPLD, 2);

  auto cvt = [&](const float* src, u16b* dst, size_t n) {
    int n4 = (int)(n / 4);
    k_f2bf4<<<dim3((unsigned)((n4 + 255) / 256)), 256, 0, stream>>>(src, (unsigned*)dst, n4);
  };
  auto zero = [&](void* dst, size_t ndw) {
    k_zero<<<dim3((unsigned)((ndw + 255) / 256)), 256, 0, stream>>>((unsigned*)dst, (int)ndw);
  };

  // ---- setup ----
  cvt(features, featbf, N_FEAT);
  cvt(W_ih, Wihbf, (size_t)3 * S * 2 * S);
  cvt(W_hh, Whhbf, (size_t)3 * S * S);
  cvt(Wq, Wqbf, (size_t)S * S);
  cvt(Wk, Wkbf, (size_t)S * S);
  k_cvt_conv<<<dim3((S * 32 + 255) / 256), 256, 0, stream>>>(conv_w, convwbf);
  k_cvt_wo<<<dim3((1008 * 1024 + 255) / 256), 256, 0, stream>>>(Wo, Wobf);
  k_emb<<<dim3((Bsz * TDEC * S + 255) / 256), 256, 0, stream>>>(input, emb, embbf);

  zero(hF0, (size_t)Bsz * S);              // f32 dwords
  zero(hbf0, (size_t)Bsz * S / 2);         // bf16 -> dwords
  zero(ctxbf, (size_t)Bsz * S / 2);
  zero(wppad, (size_t)Bsz * WPLD / 2);

  // keys = features @ Wk^T + bk
  k_keys<<<dim3((unsigned)(Bsz * TFEAT / 16), S / 16), 32, 0, stream>>>(featbf, Wkbf, bk, keysbf);

  // ---- decode loop ----
  float* hF[2]  = {hF0, hF1};
  u16b*  hbf[2] = {hbf0, hbf1};
  for (int t = 0; t < TDEC; ++t) {
    const int cur = t & 1, nxt = cur ^ 1;
    k_gru<<<dim3(Bsz / 16, S / 16), 32, 0, stream>>>(
        embbf, t, ctxbf, hbf[cur], hF[cur], Wihbf, Whhbf, b_ih, b_hh, hF[nxt], hbf[nxt]);
    k_q<<<dim3(Bsz / 16, S / 16), 32, 0, stream>>>(hbf[nxt], Wqbf, bq, qF);
    k_scores<<<dim3(Bsz, TFEAT / 16), 32, 0, stream>>>(
        wppad, convwbf, conv_b, qF, keysbf, vvec, scoresF);
    k_softmax<<<dim3(Bsz), 256, 0, stream>>>(scoresF, wsout, wbf, wppad, t);
    k_ctx<<<dim3(Bsz, 2), 128, 0, stream>>>(wbf, featbf, ctxbf);
    k_logits<<<dim3(Bsz / 16, 63), 32, 0, stream>>>(hbf[nxt], ctxbf, Wobf, bo, outs, t);
  }
}
